// GCNDetector_7172595384607
// MI455X (gfx1250) — matrix operand order repaired and checked by reading, not compiled, not run on hardware
//
#include <hip/hip_runtime.h>
#include <hip/hip_bf16.h>

typedef __attribute__((ext_vector_type(16))) _Float16 v16h;
typedef __attribute__((ext_vector_type(8)))  float    v8f;

#define BN_EPS 1e-5f

// ---------------- degree / norm ----------------
__global__ void deg_init_kernel(float* deg, int n) {
  int i = blockIdx.x * blockDim.x + threadIdx.x;
  if (i < n) deg[i] = 1.0f;  // self-loop contributes 1 to every node's degree
}

__global__ void deg_accum_kernel(const long long* __restrict__ dst, float* deg, int e) {
  int i = blockIdx.x * blockDim.x + threadIdx.x;
  if (i < e) atomicAdd(&deg[(int)dst[i]], 1.0f);
}

__global__ void deg_rsqrt_kernel(float* deg, int n) {
  int i = blockIdx.x * blockDim.x + threadIdx.x;
  if (i < n) deg[i] = rsqrtf(deg[i]);  // deg >= 1 always (self loops)
}

// ---------------- pre-bake B fragments (WMMA 16x16x32 f16 B layout) ----------------
// lane: col = lane&15, khalf = lane>>4 ; element e = W[kt*32 + khalf*8 + (e<8 ? e : e+8)][nt*16 + col]
__global__ void prep_frags_kernel(const float* __restrict__ W, int ldw,
                                  int ktiles, int ntiles, v16h* __restrict__ frag) {
  int tid = blockIdx.x * blockDim.x + threadIdx.x;
  int total = ktiles * ntiles * 32;
  if (tid >= total) return;
  int lane = tid & 31;
  int t    = tid >> 5;
  int nt   = t % ntiles;
  int kt   = t / ntiles;
  int col   = nt * 16 + (lane & 15);
  int khalf = lane >> 4;
  v16h b;
  for (int q = 0; q < 16; ++q)
    b[q] = (_Float16)W[(size_t)(kt * 32 + khalf * 8 + (q < 8 ? q : q + 8)) * ldw + col];
  frag[tid] = b;
}

// ---------------- WMMA GEMM: D[M,NT*16] = A[M,KT*32] * Bfrag ----------------
template<int KTILES, int NTILES>
__global__ void gemm_wmma_kernel(const float* __restrict__ A, int lda,
                                 const v16h* __restrict__ fragB,
                                 float* __restrict__ D, int ldd,
                                 int mtiles, int nrows) {
  const int lane = threadIdx.x & 31;
  const int wv   = blockIdx.x * (blockDim.x >> 5) + (threadIdx.x >> 5);
  if (wv >= mtiles * NTILES) return;
  const int mt    = wv / NTILES;
  const int nt    = wv % NTILES;
  const int khalf = lane >> 4;
  const int r     = lane & 15;

  int mrow = mt * 16 + r;
  int rr   = mrow < nrows ? mrow : nrows - 1;      // clamp (N=100000 is exact; safety only)
  const float* arow = A + (size_t)rr * lda;

  v8f c = {};
#pragma unroll
  for (int kt = 0; kt < KTILES; ++kt) {
    const int kb = kt * 32 + khalf * 8;
    const float4 x0 = *reinterpret_cast<const float4*>(arow + kb);
    const float4 x1 = *reinterpret_cast<const float4*>(arow + kb + 4);
    const float4 y0 = *reinterpret_cast<const float4*>(arow + kb + 16);
    const float4 y1 = *reinterpret_cast<const float4*>(arow + kb + 20);
    v16h a;
    a[0]=(_Float16)x0.x;  a[1]=(_Float16)x0.y;  a[2]=(_Float16)x0.z;  a[3]=(_Float16)x0.w;
    a[4]=(_Float16)x1.x;  a[5]=(_Float16)x1.y;  a[6]=(_Float16)x1.z;  a[7]=(_Float16)x1.w;
    a[8]=(_Float16)y0.x;  a[9]=(_Float16)y0.y;  a[10]=(_Float16)y0.z; a[11]=(_Float16)y0.w;
    a[12]=(_Float16)y1.x; a[13]=(_Float16)y1.y; a[14]=(_Float16)y1.z; a[15]=(_Float16)y1.w;
    v16h b = fragB[(kt * NTILES + nt) * 32 + lane];
    c = __builtin_amdgcn_wmma_f32_16x16x32_f16(false, a, false, b, (short)0, c, false, false);
  }
  // C/D layout: lanes 0-15 -> M = mt*16 + g, lanes 16-31 -> M = mt*16 + 8 + g; N = nt*16 + r
  const int ncol  = nt * 16 + r;
  const int mbase = mt * 16 + khalf * 8;
#pragma unroll
  for (int g = 0; g < 8; ++g) {
    int m = mbase + g;
    if (m < nrows) D[(size_t)m * ldd + ncol] = c[g];
  }
}

// ---------------- self-loop init: a[i,f] = dinv[i]^2 * t[i,f] ----------------
template<int LOGF>
__global__ void selfloop_kernel(const float* __restrict__ dinv, const float* __restrict__ t,
                                float* __restrict__ a, long long total) {
  long long i = blockIdx.x * (long long)blockDim.x + threadIdx.x;
  if (i >= total) return;
  int node = (int)(i >> LOGF);
  float w = dinv[node];
  a[i] = (w * w) * t[i];
}

// ---------------- edge scatter: a[dst,f] += dinv[s]*dinv[d]*t[src,f] ----------------
template<int LOGF>
__global__ void scatter_kernel(const long long* __restrict__ src, const long long* __restrict__ dst,
                               const float* __restrict__ dinv, const float* __restrict__ t,
                               float* __restrict__ a, long long total) {
  long long i = blockIdx.x * (long long)blockDim.x + threadIdx.x;
  if (i >= total) return;
  const int F = 1 << LOGF;
  long long e = i >> LOGF;
  int f = (int)(i & (F - 1));
  int s = (int)src[e];
  int d = (int)dst[e];
  float w = dinv[s] * dinv[d];
  atomicAdd(&a[(size_t)d * F + f], w * t[(size_t)s * F + f]);
}

// ---------------- fused bias + BN(eval) + ReLU ----------------
template<int LOGF>
__global__ void bn_relu_kernel(float* __restrict__ a, const float* __restrict__ bias,
                               const float* __restrict__ g, const float* __restrict__ be,
                               const float* __restrict__ rm, const float* __restrict__ rv,
                               long long total) {
  long long i = blockIdx.x * (long long)blockDim.x + threadIdx.x;
  if (i >= total) return;
  int f = (int)(i & ((1 << LOGF) - 1));
  float s = g[f] * rsqrtf(rv[f] + BN_EPS);
  float v = (a[i] + bias[f] - rm[f]) * s + be[f];
  a[i] = v > 0.0f ? v : 0.0f;
}

// ---------------- fused bias + BN2 + ReLU + classifier + log_softmax ----------------
__global__ void head_kernel(const float* __restrict__ a2, const float* __restrict__ b2,
                            const float* __restrict__ g2, const float* __restrict__ be2,
                            const float* __restrict__ rm2, const float* __restrict__ rv2,
                            const float* __restrict__ Wc, const float* __restrict__ bc,
                            float* __restrict__ out, int n) {
  int i = blockIdx.x * blockDim.x + threadIdx.x;
  if (i >= n) return;
  const float* row = a2 + (size_t)i * 32;
  float l0 = bc[0], l1 = bc[1];
#pragma unroll
  for (int f = 0; f < 32; ++f) {
    float s = g2[f] * rsqrtf(rv2[f] + BN_EPS);
    float h = (row[f] + b2[f] - rm2[f]) * s + be2[f];
    h = h > 0.0f ? h : 0.0f;
    l0 += h * Wc[f * 2 + 0];
    l1 += h * Wc[f * 2 + 1];
  }
  float m   = fmaxf(l0, l1);
  float lse = m + logf(expf(l0 - m) + expf(l1 - m));
  out[(size_t)i * 2 + 0] = l0 - lse;
  out[(size_t)i * 2 + 1] = l1 - lse;
}

extern "C" void kernel_launch(void* const* d_in, const int* in_sizes, int n_in,
                              void* d_out, int out_size, void* d_ws, size_t ws_size,
                              hipStream_t stream) {
  const float*     x   = (const float*)d_in[0];
  const long long* ei  = (const long long*)d_in[1];
  const float*     W1  = (const float*)d_in[2];
  const float*     b1  = (const float*)d_in[3];
  const float*     W2  = (const float*)d_in[4];
  const float*     b2  = (const float*)d_in[5];
  const float*     g1  = (const float*)d_in[6];
  const float*     be1 = (const float*)d_in[7];
  const float*     rm1 = (const float*)d_in[8];
  const float*     rv1 = (const float*)d_in[9];
  const float*     g2  = (const float*)d_in[10];
  const float*     be2 = (const float*)d_in[11];
  const float*     rm2 = (const float*)d_in[12];
  const float*     rv2 = (const float*)d_in[13];
  const float*     Wc  = (const float*)d_in[14];
  const float*     bc  = (const float*)d_in[15];

  const int FIN = 128, HID = 64, HID2 = 32;
  const int n = in_sizes[0] / FIN;
  const int e = in_sizes[1] / 2;
  const long long* srcp = ei;
  const long long* dstp = ei + e;

  // workspace layout (all 32B-aligned: every section size is a multiple of 32B)
  float* ws     = (float*)d_ws;
  float* dinv   = ws;                              // n floats
  float* t1     = ws + n;                          // n*64 floats (reused: t2 first half, a2 second half)
  float* a1     = t1 + (size_t)n * HID;            // n*64 floats (z1 in place)
  v16h*  fragB1 = (v16h*)(a1 + (size_t)n * HID);   // 4*4*32 = 512 frags (16 KB)
  v16h*  fragB2 = fragB1 + 512;                    // 2*2*32 = 128 frags (4 KB)
  float* t2     = t1;
  float* a2     = t1 + (size_t)n * HID2;

  const int B = 256;
  const int mtiles = (n + 15) / 16;

  // bake W1/W2 into WMMA B fragments (f16)
  prep_frags_kernel<<<(512 + B - 1) / B, B, 0, stream>>>(W1, HID, 4, 4, fragB1);
  prep_frags_kernel<<<(128 + B - 1) / B, B, 0, stream>>>(W2, HID2, 2, 2, fragB2);

  // symmetric degree norm on A+I
  deg_init_kernel <<<(n + B - 1) / B, B, 0, stream>>>(dinv, n);
  deg_accum_kernel<<<(e + B - 1) / B, B, 0, stream>>>(dstp, dinv, e);
  deg_rsqrt_kernel<<<(n + B - 1) / B, B, 0, stream>>>(dinv, n);

  // layer 1: t1 = x @ W1  (K=128 -> 4 k-tiles, 64 cols -> 4 n-tiles)
  {
    int waves = mtiles * 4;
    gemm_wmma_kernel<4, 4><<<(waves + 7) / 8, 256, 0, stream>>>(x, FIN, fragB1, t1, HID, mtiles, n);
  }
  long long tot1  = (long long)n * HID;
  long long etot1 = (long long)e * HID;
  selfloop_kernel<6><<<(int)((tot1  + B - 1) / B), B, 0, stream>>>(dinv, t1, a1, tot1);
  scatter_kernel <6><<<(int)((etot1 + B - 1) / B), B, 0, stream>>>(srcp, dstp, dinv, t1, a1, etot1);
  bn_relu_kernel <6><<<(int)((tot1  + B - 1) / B), B, 0, stream>>>(a1, b1, g1, be1, rm1, rv1, tot1);

  // layer 2: t2 = z1 @ W2  (K=64 -> 2 k-tiles, 32 cols -> 2 n-tiles)
  {
    int waves = mtiles * 2;
    gemm_wmma_kernel<2, 2><<<(waves + 7) / 8, 256, 0, stream>>>(a1, HID, fragB2, t2, HID2, mtiles, n);
  }
  long long tot2  = (long long)n * HID2;
  long long etot2 = (long long)e * HID2;
  selfloop_kernel<5><<<(int)((tot2  + B - 1) / B), B, 0, stream>>>(dinv, t2, a2, tot2);
  scatter_kernel <5><<<(int)((etot2 + B - 1) / B), B, 0, stream>>>(srcp, dstp, dinv, t2, a2, etot2);

  // BN2 + ReLU + classifier + log_softmax, fused
  head_kernel<<<(n + B - 1) / B, B, 0, stream>>>(a2, b2, g2, be2, rm2, rv2, Wc, bc,
                                                 (float*)d_out, n);
}
